// PropagationUnit_89713276879121
// MI455X (gfx1250) — matrix operand order repaired and checked by reading, not compiled, not run on hardware
//
#include <hip/hip_runtime.h>
#include <hip/hip_bf16.h>
#include <math.h>
#include <stdint.h>

typedef __bf16 bf16_t;
typedef __attribute__((ext_vector_type(16))) __bf16 v16bf;
typedef __attribute__((ext_vector_type(8)))  float  v8f;

#define TT  512
#define BB  256
#define II  256
#define BRR 128
#define HH  256

// Fast activations: raw v_exp_f32 / v_rcp_f32.
__device__ __forceinline__ float sigmoid_f(float x) {
  return __builtin_amdgcn_rcpf(1.0f + __builtin_amdgcn_exp2f(x * -1.44269504088896f));
}
__device__ __forceinline__ float tanh_f(float x) {
  return 1.0f - 2.0f * __builtin_amdgcn_rcpf(1.0f + __builtin_amdgcn_exp2f(x * 2.88539008177793f));
}

// ---------------------------------------------------------------------------
// Weight packing: W row-major [N][K] (i.e. W^T as a K x N B-operand) packed
// into WMMA B-fragment layout. Fragment = K32 x N16:
//   lane <16 : N = nt*16 + lane,      K = kt*32 + (e<8 ? e : e+8)      (e = 0..15)
//   lane>=16 : N = nt*16 + lane - 16, K = kt*32 + 8 + (e<8 ? e : e+8)
// ---------------------------------------------------------------------------
__global__ void pack_b(const float* __restrict__ W, bf16_t* __restrict__ dst,
                       int N, int K) {
  int tid = blockIdx.x * blockDim.x + threadIdx.x;
  int KT = K >> 5;
  int total = (N >> 4) * KT * 32;
  if (tid >= total) return;
  int lane = tid & 31;
  int frag = tid >> 5;
  int kt = frag % KT;
  int nt = frag / KT;
  int n  = nt * 16 + (lane & 15);
  int kb = kt * 32 + ((lane < 16) ? 0 : 8);
  const float* src = W + (size_t)n * K + kb;
  bf16_t* out = dst + (size_t)tid * 16;
#pragma unroll
  for (int e = 0; e < 16; ++e) out[e] = (bf16_t)src[(e < 8) ? e : e + 8];
}

__global__ void sum_bias(const float* __restrict__ a, const float* __restrict__ b,
                         const float* __restrict__ d, float* __restrict__ out, int n) {
  int i = blockIdx.x * blockDim.x + threadIdx.x;
  if (i < n) out[i] = a[i] + b[i] + d[i];
}

__global__ void cvt_f32_bf16(const float* __restrict__ src, bf16_t* __restrict__ dst,
                             int n) {
  int i = (blockIdx.x * blockDim.x + threadIdx.x) * 8;
  if (i >= n) return;
#pragma unroll
  for (int e = 0; e < 8; ++e) dst[i + e] = (bf16_t)src[i + e];
}

__global__ void zero_f32(float* __restrict__ p, int n) {
  int i = blockIdx.x * blockDim.x + threadIdx.x;
  if (i < n) p[i] = 0.0f;
}
__global__ void zero_bf16(bf16_t* __restrict__ p, int n) {
  int i = blockIdx.x * blockDim.x + threadIdx.x;
  if (i < n) p[i] = (bf16_t)0.0f;
}

// ---------------------------------------------------------------------------
// A-fragment builder (16x32 bf16, ISA 16-bit A layout).
// ---------------------------------------------------------------------------
__device__ __forceinline__ v16bf a_frag_bf16(const bf16_t* __restrict__ src, int ld,
                                             int m0, int k0, int lane) {
  int m = m0 + (lane & 15);
  const bf16_t* row = src + (size_t)m * ld + k0;
  int k1 = (lane < 16) ? 0 : 8;
  int k2 = (lane < 16) ? 16 : 24;
  v16bf a;
#pragma unroll
  for (int e = 0; e < 8; ++e) a[e] = row[k1 + e];
#pragma unroll
  for (int e = 0; e < 8; ++e) a[8 + e] = row[k2 + e];
  return a;
}

// ---------------------------------------------------------------------------
// Software-pipelined GEMM segment: acc[0..3] += A(16 x 32*KT) * B-tiles.
// A/B operands for step kt+1 are preloaded before the WMMAs of step kt so
// loads overlap matrix-core execution instead of stalling on loadcnt 0.
// ---------------------------------------------------------------------------
template <int KT>
__device__ __forceinline__ void gemm_seg(const bf16_t* __restrict__ Asrc, int ld,
                                         int m0, const v16bf* __restrict__ W,
                                         int ntBase, int lane, v8f* acc) {
  v16bf a = a_frag_bf16(Asrc, ld, m0, 0, lane);
  v16bf b[4];
#pragma unroll
  for (int j = 0; j < 4; ++j)
    b[j] = W[(size_t)((ntBase + j) * KT) * 32 + lane];
#pragma unroll
  for (int kt = 0; kt < KT; ++kt) {
    v16bf an = a;
    v16bf bn[4] = {b[0], b[1], b[2], b[3]};
    if (kt + 1 < KT) {
      an = a_frag_bf16(Asrc, ld, m0, (kt + 1) * 32, lane);
#pragma unroll
      for (int j = 0; j < 4; ++j)
        bn[j] = W[(size_t)((ntBase + j) * KT + kt + 1) * 32 + lane];
    }
#pragma unroll
    for (int j = 0; j < 4; ++j)
      acc[j] = __builtin_amdgcn_wmma_f32_16x16x32_bf16(
          false, a, false, b[j], (short)0, acc[j], false, false);
    a = an;
#pragma unroll
    for (int j = 0; j < 4; ++j) b[j] = bn[j];
  }
}

// ---------------------------------------------------------------------------
// Phase 1 (time-parallel): bhf = x @ W_f^T + b_f for all T*B rows.
// ---------------------------------------------------------------------------
__global__ void precompute_gates(const bf16_t* __restrict__ xb,   // [R, II] bf16
                                 const float* __restrict__ brg,   // [R, BRR]
                                 const bf16_t* __restrict__ WfP,
                                 const float* __restrict__ b_f,
                                 bf16_t* __restrict__ fgate,      // [R, HH]
                                 bf16_t* __restrict__ bg)         // [R, BRR]
{
  int lane = threadIdx.x & 31;
  int wave = blockIdx.x * (blockDim.x >> 5) + (threadIdx.x >> 5);
  const int STRIPS = (HH + BRR) / 64;   // 6
  int strip = wave % STRIPS;
  int m0 = (wave / STRIPS) * 16;
  int n0 = strip * 64;

  v8f acc[4] = {};
  gemm_seg<II / 32>(xb, II, m0, (const v16bf*)WfP, n0 >> 4, lane, acc);

#pragma unroll
  for (int j = 0; j < 4; ++j) {
    int col = n0 + j * 16 + (lane & 15);
    float bias = b_f[col];
#pragma unroll
    for (int r = 0; r < 8; ++r) {
      int m = m0 + ((lane < 16) ? r : 8 + r);
      float s = sigmoid_f(acc[j][r] + bias);
      if (col < HH) {
        fgate[(size_t)m * HH + col] = (bf16_t)s;
      } else {
        int cb = col - HH;
        bg[(size_t)m * BRR + cb] = (bf16_t)(s * brg[(size_t)m * BRR + cb]);
      }
    }
  }
}

// ---------------------------------------------------------------------------
// Phase 2 (serial over T): one launch per step. Workgroup = 4 waves; wave g
// computes gate chunk g of a 16-row x 64-hcol tile:
//   gates = xb@Wxh^T + hg@Whh^T + bg@Wbh^T + bias_sum      (K = 640 total)
// hg (= fgate[t]*h[t-1]) was produced by the PREVIOUS step's epilogue into a
// parity double-buffer; this step's epilogue emits hg for step t+1.
// ---------------------------------------------------------------------------
__global__ void lstm_step(const bf16_t* __restrict__ xb_t,    // [BB, II]
                          const bf16_t* __restrict__ bg_t,    // [BB, BRR]
                          const bf16_t* __restrict__ hg_in,   // [BB, HH]
                          const bf16_t* __restrict__ fg_next, // [BB, HH] (fgate[t+1])
                          float* __restrict__ c,              // [BB, HH]
                          float* __restrict__ h_out,          // [BB, HH] (out[t])
                          bf16_t* __restrict__ hg_out,        // [BB, HH]
                          const bf16_t* __restrict__ WxhP,
                          const bf16_t* __restrict__ WhhP,
                          const bf16_t* __restrict__ WbhP,
                          const float* __restrict__ bias_sum) // [4H]
{
  __shared__ float gtile[4][16][65];
  int lane = threadIdx.x & 31;
  int g    = threadIdx.x >> 5;         // 0=f 1=i 2=g 3=o
  int m0   = blockIdx.y * 16;
  int hc0  = blockIdx.x * 64;
  int N0   = g * HH + hc0;

  v8f acc[4] = {};
  gemm_seg<II / 32>(xb_t, II, m0, (const v16bf*)WxhP, N0 >> 4, lane, acc);
  gemm_seg<HH / 32>(hg_in, HH, m0, (const v16bf*)WhhP, N0 >> 4, lane, acc);
  gemm_seg<BRR / 32>(bg_t, BRR, m0, (const v16bf*)WbhP, N0 >> 4, lane, acc);

  // Stage gate tiles into LDS (C layout: lane<16 -> M=r, lane>=16 -> M=8+r).
#pragma unroll
  for (int j = 0; j < 4; ++j) {
    int n = N0 + j * 16 + (lane & 15);
    float bias = bias_sum[n];
    int cl = j * 16 + (lane & 15);
#pragma unroll
    for (int r = 0; r < 8; ++r) {
      int ml = (lane < 16) ? r : 8 + r;
      gtile[g][ml][cl] = acc[j][r] + bias;
    }
  }
  __syncthreads();

  // Fused elementwise update + production of next step's recurrent operand.
#pragma unroll
  for (int e = 0; e < 8; ++e) {
    int li = threadIdx.x + 128 * e;
    int ml = li >> 6, cl = li & 63;
    float fv = gtile[0][ml][cl];
    float iv = gtile[1][ml][cl];
    float gv = gtile[2][ml][cl];
    float ov = gtile[3][ml][cl];
    size_t idx = (size_t)(m0 + ml) * HH + hc0 + cl;
    float cn = c[idx] * sigmoid_f(fv) + sigmoid_f(iv) * tanh_f(gv);
    float hn = sigmoid_f(ov) * tanh_f(cn);
    c[idx] = cn;
    h_out[idx] = hn;
    hg_out[idx] = (bf16_t)((float)fg_next[idx] * hn);
  }
}

__global__ void tail_copy(const float* __restrict__ hT, const float* __restrict__ cT,
                          float* __restrict__ dst, int n) {
  int i = blockIdx.x * blockDim.x + threadIdx.x;
  if (i < n) { dst[i] = hT[i]; dst[n + i] = cT[i]; }
}

// ---------------------------------------------------------------------------
extern "C" void kernel_launch(void* const* d_in, const int* in_sizes, int n_in,
                              void* d_out, int out_size, void* d_ws, size_t ws_size,
                              hipStream_t stream) {
  (void)in_sizes; (void)n_in; (void)out_size; (void)ws_size;
  const float* x    = (const float*)d_in[0];  // [T,B,I]
  const float* brg  = (const float*)d_in[1];  // [T,B,BR]
  const float* W_f  = (const float*)d_in[2];  // [H+BR, I]
  const float* b_f  = (const float*)d_in[3];
  const float* W_xh = (const float*)d_in[4];  // [4H, I]
  const float* b_xh = (const float*)d_in[5];
  const float* W_bh = (const float*)d_in[6];  // [4H, BR]
  const float* b_bh = (const float*)d_in[7];
  const float* W_hh = (const float*)d_in[8];  // [4H, H]
  const float* b_hh = (const float*)d_in[9];
  float* out = (float*)d_out;                 // [T,B,H] ++ hT[B,H] ++ cT[B,H]

  uint8_t* ws = (uint8_t*)d_ws;
  size_t off = 0;
  auto take = [&](size_t bytes) -> void* {
    void* p = ws + off;
    off += (bytes + 255) & ~(size_t)255;
    return p;
  };
  bf16_t* WfP   = (bf16_t*)take((size_t)((HH + BRR) / 16) * (II / 32) * 512 * 2);
  bf16_t* WxhP  = (bf16_t*)take((size_t)((4 * HH) / 16) * (II / 32) * 512 * 2);
  bf16_t* WbhP  = (bf16_t*)take((size_t)((4 * HH) / 16) * (BRR / 32) * 512 * 2);
  bf16_t* WhhP  = (bf16_t*)take((size_t)((4 * HH) / 16) * (HH / 32) * 512 * 2);
  float*  biasS = (float*)take((size_t)4 * HH * 4);
  bf16_t* xb    = (bf16_t*)take((size_t)TT * BB * II * 2);
  bf16_t* fgate = (bf16_t*)take((size_t)TT * BB * HH * 2);
  bf16_t* bgbuf = (bf16_t*)take((size_t)TT * BB * BRR * 2);
  bf16_t* hg0   = (bf16_t*)take((size_t)BB * HH * 2);
  bf16_t* hg1   = (bf16_t*)take((size_t)BB * HH * 2);
  float*  cbuf  = (float*)take((size_t)BB * HH * 4);

  // Phase 0: weight packing, bias fusion, x -> bf16, state init.
  {
    int th;
    th = ((HH + BRR) / 16) * (II / 32) * 32;
    pack_b<<<(th + 255) / 256, 256, 0, stream>>>(W_f, WfP, HH + BRR, II);
    th = ((4 * HH) / 16) * (II / 32) * 32;
    pack_b<<<(th + 255) / 256, 256, 0, stream>>>(W_xh, WxhP, 4 * HH, II);
    th = ((4 * HH) / 16) * (BRR / 32) * 32;
    pack_b<<<(th + 255) / 256, 256, 0, stream>>>(W_bh, WbhP, 4 * HH, BRR);
    th = ((4 * HH) / 16) * (HH / 32) * 32;
    pack_b<<<(th + 255) / 256, 256, 0, stream>>>(W_hh, WhhP, 4 * HH, HH);
  }
  sum_bias<<<(4 * HH + 255) / 256, 256, 0, stream>>>(b_xh, b_hh, b_bh, biasS, 4 * HH);
  {
    int n = TT * BB * II;
    cvt_f32_bf16<<<(n / 8 + 255) / 256, 256, 0, stream>>>(x, xb, n);
  }
  zero_bf16<<<(BB * HH + 255) / 256, 256, 0, stream>>>(hg0, BB * HH);  // h0 == 0
  zero_f32<<<(BB * HH + 255) / 256, 256, 0, stream>>>(cbuf, BB * HH);

  // Phase 1: one big batched WMMA GEMM over all T (off the critical path).
  {
    int R = TT * BB;
    int waves = (R / 16) * ((HH + BRR) / 64);  // 49152
    precompute_gates<<<waves / 8, 256, 0, stream>>>(xb, brg, WfP, b_f, fgate, bgbuf);
  }

  // Phase 2: 512 stream-ordered recurrent steps.
  for (int t = 0; t < TT; ++t) {
    const bf16_t* xb_t = xb    + (size_t)t * BB * II;
    const bf16_t* bg_t = bgbuf + (size_t)t * BB * BRR;
    const bf16_t* hgi  = (t & 1) ? hg1 : hg0;
    bf16_t*       hgo  = (t & 1) ? hg0 : hg1;
    int tn = (t + 1 < TT) ? (t + 1) : (TT - 1);   // value unused at t = T-1
    const bf16_t* fgn  = fgate + (size_t)tn * BB * HH;
    float*        h_o  = out + (size_t)t * BB * HH;
    dim3 grid(HH / 64, BB / 16);
    lstm_step<<<grid, 128, 0, stream>>>(xb_t, bg_t, hgi, fgn, cbuf, h_o, hgo,
                                        WxhP, WhhP, WbhP, biasS);
  }

  // Append hT (= out[T-1]) and cT to the output buffer.
  tail_copy<<<(BB * HH + 255) / 256, 256, 0, stream>>>(
      out + (size_t)(TT - 1) * BB * HH, cbuf, out + (size_t)TT * BB * HH, BB * HH);
}
